// Encoder_19086834663950
// MI455X (gfx1250) — compile-verified, matrix operand order checked by reading, not run on hardware
//
#include <hip/hip_runtime.h>
#include <hip/hip_bf16.h>
#include <math.h>

// Problem constants (match reference)
#define DD 4096   // feature dim D
#define RB 4096   // batch rows B
#define NF 10     // number of flows
#define NH 3      // hidden layers
#define HD 8      // hidden width H

typedef __attribute__((ext_vector_type(16))) _Float16 v16h;
typedef __attribute__((ext_vector_type(8)))  float    v8f;
typedef __attribute__((ext_vector_type(4)))  int      v4i;

// ---------------------------------------------------------------------------
// Kernel 1: per-row tiny recurrence. MADE masks make h depend only on x[0:8],
// and x[0:8] evolves depending only on itself. One thread per row runs the
// 10-flow recurrence (f32) and stores h[f][0:8] as f16 for the WMMA kernel.
// Weight loads are uniform across threads -> scalar/broadcast loads.
// ---------------------------------------------------------------------------
__global__ __launch_bounds__(256) void made_recurrence(
    const float* __restrict__ X, const float* __restrict__ W0,
    const float* __restrict__ b0, const float* __restrict__ Wh,
    const float* __restrict__ bh, const float* __restrict__ Wo,
    const float* __restrict__ bo, _Float16* __restrict__ Hws)
{
    const int row = blockIdx.x * blockDim.x + threadIdx.x;
    if (row >= RB) return;

    float x8[HD];
#pragma unroll
    for (int j = 0; j < HD; ++j) x8[j] = X[(size_t)row * DD + j];

    for (int f = 0; f < NF; ++f) {
        const float* w0f = W0 + (size_t)f * HD * DD;
        const float* b0f = b0 + f * HD;
        float h[HD];
#pragma unroll
        for (int j = 0; j < HD; ++j) {                 // m0 keeps i <= j
            float s = b0f[j];
#pragma unroll
            for (int i = 0; i <= j; ++i) s += w0f[(size_t)j * DD + i] * x8[i];
            h[j] = fmaxf(s, 0.0f);
        }
        for (int l = 0; l < NH; ++l) {                 // mh keeps i <= j
            const float* whf = Wh + (size_t)(f * NH + l) * HD * HD;
            const float* bhf = bh + (f * NH + l) * HD;
            float nh[HD];
#pragma unroll
            for (int j = 0; j < HD; ++j) {
                float s = bhf[j];
#pragma unroll
                for (int i = 0; i <= j; ++i) s += whf[j * HD + i] * h[i];
                nh[j] = fmaxf(s, 0.0f);
            }
#pragma unroll
            for (int j = 0; j < HD; ++j) h[j] = nh[j];
        }
#pragma unroll
        for (int j = 0; j < HD; ++j)                   // stash h as f16
            Hws[(size_t)row * (NF * HD) + f * HD + j] = (_Float16)h[j];

        const float* wof = Wo + (size_t)f * 2 * DD * HD;
        const float* bof = bo + (size_t)f * 2 * DD;
#pragma unroll
        for (int i = 0; i < HD; ++i) {                 // mo keeps j < i
            float m  = bof[i];
            float ls = bof[DD + i];
#pragma unroll
            for (int j = 0; j < HD; ++j) {
                if (j < i) {
                    m  += wof[(size_t)i * HD + j] * h[j];
                    ls += wof[(size_t)(DD + i) * HD + j] * h[j];
                }
            }
            ls = fminf(fmaxf(ls, -5.0f), 3.0f);
            x8[i] = __expf(ls) * x8[i] + m;
        }
    }
}

// ---------------------------------------------------------------------------
// Kernel 2: per-element flow application via f16 WMMA (f32 accumulate).
// The block's 4 waves share one 16-column tile. The 20 packed B operands
// (10 flows x {mean,logs}; mask j<col folded in; K=8..31 zero-padded) are
// staged once in LDS (20 KB) and re-read per flow with ds_load_b128 — an
// opaque asm token keeps LICM from hoisting them back into registers. Each
// wave processes 2 row tiles per B read. Per flow/tile: ONE
// v_wmma_f32_16x16x32_f16 each for Mean and LogS with C = 0 (rematerializable
// / inline-0 seed, no spillable live range); biases are added in the
// elementwise tail: x = exp(clip(ls+bL,-5,3))*x + (mean+bM).
//
// Layouts (ISA 7.12.2, 16-bit): A/B nonzero content lives in halfs 0..7 of
// lanes 0-15 (K=0..7); zero-padding all other slots is valid under every
// documented 16-bit layout variant. Wave-uniform control flow around WMMA.
// ---------------------------------------------------------------------------
__global__ __launch_bounds__(128, 4) void made_apply_wmma(
    const float* __restrict__ X, const float* __restrict__ Wo,
    const float* __restrict__ bo, const _Float16* __restrict__ Hws,
    float* __restrict__ Y)
{
    __shared__ v16h WLDS[NF * 2 * 32];   // 20 operands x 32 lanes x 32B = 20 KB

    const int tid  = threadIdx.x;
    const int lane = tid & 31;
    const int wave = tid >> 5;
    const int c0   = blockIdx.x * 16;
    const int col  = c0 + (lane & 15);

    // ---- stage packed f16 B operands into LDS (block-shared) ----
    for (int f = wave; f < NF; f += 4) {
        const float* wm = Wo + ((size_t)f * 2 * DD + col) * HD;   // mean row
        const float* wl = wm + (size_t)DD * HD;                   // log_s row
        v16h m = {};
        v16h l = {};
        if (lane < 16) {
#pragma unroll
            for (int j = 0; j < HD; ++j) {
                if (j < col) { m[j] = (_Float16)wm[j]; l[j] = (_Float16)wl[j]; }
            }
        }
        WLDS[(f * 2 + 0) * 32 + lane] = m;
        WLDS[(f * 2 + 1) * 32 + lane] = l;
    }
    __syncthreads();

    const int loMask = (lane < 16) ? -1 : 0;   // lanes 0-15 carry K=0..7

    float bsM[NF], bsL[NF];
#pragma unroll
    for (int f = 0; f < NF; ++f) {
        bsM[f] = bo[(size_t)f * 2 * DD + col];
        bsL[f] = bo[(size_t)f * 2 * DD + DD + col];
    }

    const v8f zc = {};   // zero C seed (rematerializable, no spill pressure)

    // 128 pairs of 16-row tiles; each wave takes 4 pairs (stride 32)
    for (int p = blockIdx.y * 4 + wave; p < RB / 32; p += gridDim.y * 4) {
        const int r0  = p * 32;
        const int rb0 = r0 + ((lane >> 4) * 8);    // C-layout row base, tile 0
        const int rb1 = rb0 + 16;                  // tile 1

        const float* xp0 = X + (size_t)rb0 * DD + col;
        const float* xp1 = X + (size_t)rb1 * DD + col;
        v8f x0, x1;
#pragma unroll
        for (int e = 0; e < 8; ++e) { x0[e] = xp0[e * DD]; x1[e] = xp1[e * DD]; }

        const _Float16* h0 = Hws + (size_t)(r0 + (lane & 15)) * (NF * HD);
        const _Float16* h1 = h0 + 16 * (NF * HD);

        int opq = 0;
        asm volatile("" : "+v"(opq));   // opaque 0: keep ds reads in this loop

#pragma unroll
        for (int f = 0; f < NF; ++f) {
            v16h Bm = WLDS[(f * 2 + 0) * 32 + lane + opq];
            v16h Bl = WLDS[(f * 2 + 1) * 32 + lane + opq];

#pragma unroll
            for (int t = 0; t < 2; ++t) {
                const _Float16* hr_p = (t == 0) ? h0 : h1;
                // Build A branchlessly: halfs 0..7 = h[f][0..7] on lanes 0-15.
                v4i hr = *(const v4i*)(hr_p + f * HD);
                union { v16h h; int i[8]; } ua;
                ua.i[0] = hr.x & loMask;
                ua.i[1] = hr.y & loMask;
                ua.i[2] = hr.z & loMask;
                ua.i[3] = hr.w & loMask;
                ua.i[4] = 0; ua.i[5] = 0; ua.i[6] = 0; ua.i[7] = 0;

                v8f mC = __builtin_amdgcn_wmma_f32_16x16x32_f16(
                             false, ua.h, false, Bm, (short)0, zc, false, false);
                v8f lC = __builtin_amdgcn_wmma_f32_16x16x32_f16(
                             false, ua.h, false, Bl, (short)0, zc, false, false);

                v8f& x = (t == 0) ? x0 : x1;
#pragma unroll
                for (int e = 0; e < 8; ++e) {
                    float ls = fminf(fmaxf(lC[e] + bsL[f], -5.0f), 3.0f);
                    x[e] = __expf(ls) * x[e] + (mC[e] + bsM[f]);
                }
            }
        }

        float* yp0 = Y + (size_t)rb0 * DD + col;
        float* yp1 = Y + (size_t)rb1 * DD + col;
#pragma unroll
        for (int e = 0; e < 8; ++e) { yp0[e * DD] = x0[e]; yp1[e * DD] = x1[e]; }
    }
}

// ---------------------------------------------------------------------------
extern "C" void kernel_launch(void* const* d_in, const int* in_sizes, int n_in,
                              void* d_out, int out_size, void* d_ws, size_t ws_size,
                              hipStream_t stream) {
    const float* X  = (const float*)d_in[0];
    const float* W0 = (const float*)d_in[1];
    const float* b0 = (const float*)d_in[2];
    const float* Wh = (const float*)d_in[3];
    const float* bh = (const float*)d_in[4];
    const float* Wo = (const float*)d_in[5];
    const float* bo = (const float*)d_in[6];
    float* Y        = (float*)d_out;
    _Float16* Hws   = (_Float16*)d_ws;   // RB * NF * HD halfs = 640 KB

    made_recurrence<<<RB / 256, 256, 0, stream>>>(X, W0, b0, Wh, bh, Wo, bo, Hws);

    dim3 grid(DD / 16, 8);   // 256 column tiles x 8 row-tile-pair groups
    made_apply_wmma<<<grid, 128, 0, stream>>>(X, Wo, bo, Hws, Y);
}
